// RNNModelScratch_71554155152160
// MI455X (gfx1250) — compile-verified, matrix operand order checked by reading, not run on hardware
//
#include <hip/hip_runtime.h>
#include <hip/hip_bf16.h>
#include <math.h>
#include <stdint.h>

#define VOCAB  4096
#define HIDDEN 1024
#define BATCH  64
#define SEQ    512
#define TILE_K 64            // phase-2 LDS staging depth (per buffer)

typedef __attribute__((ext_vector_type(16))) __bf16 v16bf;
typedef __attribute__((ext_vector_type(8)))  float  v8f;

union FragU { uint4 u[2]; v16bf v; };

// float -> bf16 (round to nearest even)
__device__ __forceinline__ unsigned short f2bf(float f) {
    unsigned int u = __float_as_uint(f);
    u = (u + 0x7FFFu + ((u >> 16) & 1u)) >> 16;
    return (unsigned short)u;
}

// 16x32 bf16 WMMA fragment from a row-major (ld-stride) bf16 matrix.
// Per-lane: row = base_row + (lane&15); K runs [k0+8h,+8) and [k0+16+8h,+8).
__device__ __forceinline__ v16bf load_frag(const unsigned short* base, int row,
                                           int ld, int k0) {
    int lane = threadIdx.x & 31;
    const unsigned short* p =
        base + (size_t)(row + (lane & 15)) * ld + k0 + ((lane >> 4) << 3);
    FragU f;
    f.u[0] = *(const uint4*)p;
    f.u[1] = *(const uint4*)(p + 16);
    return f.v;
}

// Same fragment pattern out of an LDS tile with row stride TILE_K.
__device__ __forceinline__ v16bf load_frag_lds(const unsigned short* base,
                                               int row, int kk) {
    int lane = threadIdx.x & 31;
    const unsigned short* p =
        base + (row + (lane & 15)) * TILE_K + kk + ((lane >> 4) << 3);
    FragU f;
    f.u[0] = *(const uint4*)p;
    f.u[1] = *(const uint4*)(p + 16);
    return f.v;
}

__device__ __forceinline__ v8f wmma_bf16(v16bf a, v16bf b, v8f c) {
    return __builtin_amdgcn_wmma_f32_16x16x32_bf16(
        false, a, false, b, (short)0, c, false, false);
}

// ---------------------------------------------------------------------------
// Pre-pass: convert f32 -> bf16 (optionally transposed)
// ---------------------------------------------------------------------------
__global__ void k_cvt_transpose(const float* __restrict__ src,
                                unsigned short* __restrict__ dst,
                                int K, int N) {      // src [K][N] -> dst [N][K]
    size_t i = (size_t)blockIdx.x * blockDim.x + threadIdx.x;
    size_t total = (size_t)K * N;
    if (i >= total) return;
    int k = (int)(i / N), n = (int)(i % N);
    dst[(size_t)n * K + k] = f2bf(src[i]);
}

__global__ void k_cvt(const float* __restrict__ src,
                      unsigned short* __restrict__ dst, int total) {
    int i = blockIdx.x * blockDim.x + threadIdx.x;
    if (i < total) dst[i] = f2bf(src[i]);
}

// ---------------------------------------------------------------------------
// Phase 1: one recurrent step.  H_new = tanh(W_xh[x_t] + H_prev@W_hh + b_h)
// Grid: 64 blocks x 128 threads; wave w -> M rows [16w,+16), N tile [16*bx,+16).
// Serial/latency-bound phase; direct global feeds, fully unrolled K pipeline.
// ---------------------------------------------------------------------------
__global__ void k_rnn_step(const unsigned short* __restrict__ Hprev,   // [64][1024] bf16
                           const unsigned short* __restrict__ WhhT,    // [1024][1024] bf16 (N-major)
                           const float* __restrict__ W_xh,             // [4096][1024] f32
                           const float* __restrict__ b_h,              // [1024]
                           const int*   __restrict__ X,                // [64][512]
                           int t,
                           unsigned short* __restrict__ Hout,          // [64][1024] bf16
                           float* __restrict__ Hfinal) {               // f32 or nullptr
    int wave = threadIdx.x >> 5;
    int lane = threadIdx.x & 31;
    int h = lane >> 4, nn = lane & 15;
    int m0 = wave << 4;
    int n0 = blockIdx.x << 4;

    v8f acc = {};
    for (int k0 = 0; k0 < HIDDEN; k0 += 32) {
        v16bf a = load_frag(Hprev, m0, HIDDEN, k0);
        v16bf b = load_frag(WhhT,  n0, HIDDEN, k0);
        acc = wmma_bf16(a, b, acc);
    }

    int n = n0 + nn;
    float bh = b_h[n];
#pragma unroll
    for (int r = 0; r < 8; ++r) {
        int m = m0 + r + (h << 3);                 // batch row 0..63
        int tok = X[m * SEQ + t];
        float v = acc[r] + W_xh[(size_t)tok * HIDDEN + n] + bh;
        v = tanhf(v);
        Hout[(size_t)m * HIDDEN + n] = f2bf(v);
        if (Hfinal) Hfinal[(size_t)m * HIDDEN + n] = v;
    }
}

// ---------------------------------------------------------------------------
// Phase 2: out[32768,4096] = Hhist @ W_hq + b_q  (bf16 WMMA, f32 accumulate)
// Block = 128 threads (4 waves) -> 128(M) x 64(N); wave w owns rows [32w,+32),
// 8 accumulators (2M x 4N tiles of 16x16).
// B tile (64K x 64N = 8 KB) shared by all 4 waves: double-buffered into LDS
// with global_load_async_to_lds_b128 (ASYNCcnt), read back as ds_load_b128
// fragments.  One s_wait_asynccnt + barrier per 64-wide K chunk (16 WMMAs per wave of
// work between barriers).  A feeds straight from global (rows per-wave).
// Logits stores are non-temporal: 512 MB written once, never re-read -> keep
// A (64 MB) and B (8 MB) resident in the 192 MB L2.
// ---------------------------------------------------------------------------
__global__ void k_out_gemm(const unsigned short* __restrict__ A,   // [32768][1024] bf16
                           const unsigned short* __restrict__ B,   // [4096][1024] bf16 (W_hq^T)
                           const float* __restrict__ b_q,          // [4096]
                           float* __restrict__ out) {              // [32768][4096]
    __shared__ unsigned short ldsB[2][64 * TILE_K];                // 2 x 8 KB

    int wave = threadIdx.x >> 5;         // 0..3
    int lane = threadIdx.x & 31;
    int h = lane >> 4, nn = lane & 15;
    int m0 = blockIdx.y * 128 + wave * 32;
    int n0 = blockIdx.x * 64;

    // async staging: wave w moves rows [16w,+16) of the 64-row B tile;
    // each op moves 4 rows x 128B (32 lanes x b128), 4 ops per wave per chunk.
    int srow = lane >> 3;                // 0..3
    int scol = (lane & 7) * 16;          // byte offset within a 128B row

    auto stage = [&](int k0, int buf) {
#pragma unroll
        for (int i = 0; i < 4; ++i) {
            int row = wave * 16 + i * 4 + srow;                      // 0..63
            unsigned long long ga = (unsigned long long)(uintptr_t)(
                B + (size_t)(n0 + row) * HIDDEN + k0) + (unsigned)scol;
            unsigned la = (unsigned)(uintptr_t)&ldsB[buf][row * TILE_K]
                          + (unsigned)scol;
            asm volatile("global_load_async_to_lds_b128 %0, %1, off"
                         :: "v"(la), "v"(ga) : "memory");
        }
    };

    v8f acc[2][4] = {};

    stage(0, 0);
    asm volatile("s_wait_asynccnt 0x0" ::: "memory");
    __syncthreads();

    for (int k0 = 0; k0 < HIDDEN; k0 += TILE_K) {
        int buf = (k0 / TILE_K) & 1;
        bool more = (k0 + TILE_K) < HIDDEN;
        if (more) stage(k0 + TILE_K, buf ^ 1);       // overlap with compute

#pragma unroll
        for (int kk = 0; kk < TILE_K; kk += 32) {
            v16bf a0 = load_frag(A, m0,      HIDDEN, k0 + kk);
            v16bf a1 = load_frag(A, m0 + 16, HIDDEN, k0 + kk);
#pragma unroll
            for (int j = 0; j < 4; ++j) {
                v16bf b = load_frag_lds(&ldsB[buf][0], j * 16, kk);
                acc[0][j] = wmma_bf16(a0, b, acc[0][j]);
                acc[1][j] = wmma_bf16(a1, b, acc[1][j]);
            }
        }

        if (more) {
            asm volatile("s_wait_asynccnt 0x0" ::: "memory");
            __syncthreads();
        }
    }

#pragma unroll
    for (int j = 0; j < 4; ++j) {
        int n = n0 + j * 16 + nn;
        float bq = b_q[n];
#pragma unroll
        for (int mm = 0; mm < 2; ++mm) {
#pragma unroll
            for (int r = 0; r < 8; ++r) {
                int m = m0 + mm * 16 + r + (h << 3);
                __builtin_nontemporal_store(acc[mm][j][r] + bq,
                                            &out[(size_t)m * VOCAB + n]);
            }
        }
    }
}

// ---------------------------------------------------------------------------
extern "C" void kernel_launch(void* const* d_in, const int* in_sizes, int n_in,
                              void* d_out, int out_size, void* d_ws, size_t ws_size,
                              hipStream_t stream) {
    const int*   X    = (const int*)d_in[0];
    const float* H0   = (const float*)d_in[1];
    const float* W_xh = (const float*)d_in[2];
    const float* W_hh = (const float*)d_in[3];
    const float* b_h  = (const float*)d_in[4];
    const float* W_hq = (const float*)d_in[5];
    const float* b_q  = (const float*)d_in[6];

    float* out    = (float*)d_out;                                  // [32768][4096]
    float* Hfinal = out + (size_t)SEQ * BATCH * VOCAB;              // [64][1024]

    // workspace layout (bf16 = unsigned short units)
    unsigned short* W_hhT = (unsigned short*)d_ws;                  // 1024*1024
    unsigned short* W_hqT = W_hhT + (size_t)HIDDEN * HIDDEN;        // 4096*1024
    unsigned short* Hinit = W_hqT + (size_t)VOCAB * HIDDEN;         // 64*1024
    unsigned short* Hhist = Hinit + (size_t)BATCH * HIDDEN;         // 512*64*1024

    // --- pre-pass: bf16 conversion (+ transpose of weights) ---
    {
        size_t n1 = (size_t)HIDDEN * HIDDEN;
        k_cvt_transpose<<<(unsigned)((n1 + 255) / 256), 256, 0, stream>>>(
            W_hh, W_hhT, HIDDEN, HIDDEN);
        size_t n2 = (size_t)HIDDEN * VOCAB;
        k_cvt_transpose<<<(unsigned)((n2 + 255) / 256), 256, 0, stream>>>(
            W_hq, W_hqT, HIDDEN, VOCAB);
        int n3 = BATCH * HIDDEN;
        k_cvt<<<(n3 + 255) / 256, 256, 0, stream>>>(H0, Hinit, n3);
    }

    // --- phase 1: 512 dependent recurrent steps ---
    for (int t = 0; t < SEQ; ++t) {
        const unsigned short* Hprev =
            (t == 0) ? Hinit : (Hhist + (size_t)(t - 1) * BATCH * HIDDEN);
        unsigned short* Hout = Hhist + (size_t)t * BATCH * HIDDEN;
        float* hf = (t == SEQ - 1) ? Hfinal : nullptr;
        k_rnn_step<<<HIDDEN / 16, 128, 0, stream>>>(
            Hprev, W_hhT, W_xh, b_h, X, t, Hout, hf);
    }

    // --- phase 2: one big batched projection GEMM ---
    dim3 grid(VOCAB / 64, (SEQ * BATCH) / 128);
    k_out_gemm<<<grid, 128, 0, stream>>>(Hhist, W_hqT, b_q, out);
}
